// ExpressionAttentionLayer_50543175139787
// MI455X (gfx1250) — matrix operand order repaired and checked by reading, not compiled, not run on hardware
//
#include <hip/hip_runtime.h>

#define B_ 8
#define G_ 2048
#define D_ 64

typedef __attribute__((ext_vector_type(16))) _Float16 v16h;
typedef __attribute__((ext_vector_type(8)))  _Float16 v8h;
typedef __attribute__((ext_vector_type(2)))  __fp16   v2h;   // matches cvt_pkrtz return
typedef __attribute__((ext_vector_type(8)))  float    v8f;

__device__ __forceinline__ v16h load16(const _Float16* lo_p, const _Float16* hi_p) {
    v8h lo = *(const v8h*)lo_p;
    v8h hi = *(const v8h*)hi_p;
    return __builtin_shufflevector(lo, hi, 0,1,2,3,4,5,6,7,8,9,10,11,12,13,14,15);
}

// DPP16 lane move within 16-lane rows (wave32): pure VALU, no LDS traffic.
template <int CTRL>
__device__ __forceinline__ float dpp_mov(float x) {
    int xi = __builtin_bit_cast(int, x);
    int r  = __builtin_amdgcn_update_dpp(xi, xi, CTRL, 0xF, 0xF, true);
    return __builtin_bit_cast(float, r);
}
// Full 16-lane reductions via butterfly: quad xor1, quad xor2, half-mirror, mirror.
__device__ __forceinline__ float red_max16(float x) {
    x = fmaxf(x, dpp_mov<0xB1>(x));   // quad_perm [1,0,3,2]
    x = fmaxf(x, dpp_mov<0x4E>(x));   // quad_perm [2,3,0,1]
    x = fmaxf(x, dpp_mov<0x141>(x));  // row_half_mirror
    x = fmaxf(x, dpp_mov<0x140>(x));  // row_mirror
    return x;
}
__device__ __forceinline__ float red_sum16(float x) {
    x += dpp_mov<0xB1>(x);
    x += dpp_mov<0x4E>(x);
    x += dpp_mov<0x141>(x);
    x += dpp_mov<0x140>(x);
    return x;
}

#define WMMA_F16(A, Bm, C) \
    __builtin_amdgcn_wmma_f32_16x16x32_f16(false, (A), false, (Bm), (short)0, (C), false, false)

// ---------------------------------------------------------------------------
// Fused projection: out[row, d] = (concat(gene,expr)[row] . W[d, :]) + bias[d]
// ---------------------------------------------------------------------------
__global__ __launch_bounds__(256) void proj_kernel(
    const float* __restrict__ Xg, const float* __restrict__ Xe,
    const float* __restrict__ W,  const float* __restrict__ bias,
    _Float16* __restrict__ out, float scale)
{
    __shared__ float Ws[D_ * 2 * D_];           // 32 KB
    for (int i = threadIdx.x; i < D_ * 2 * D_; i += 256) Ws[i] = W[i];
    __syncthreads();

    int idx = blockIdx.x * 256 + threadIdx.x;   // over B*G*D
    int row = idx >> 6;
    int d   = idx & 63;
    const float* gr = Xg + (size_t)row * D_;
    const float* er = Xe + (size_t)row * D_;
    const float* w  = Ws + d * 2 * D_;
    float acc = bias[d];
    #pragma unroll
    for (int k = 0; k < D_; ++k) acc = fmaf(gr[k], w[k], acc);
    #pragma unroll
    for (int k = 0; k < D_; ++k) acc = fmaf(er[k], w[D_ + k], acc);
    out[idx] = (_Float16)(acc * scale);
}

// ---------------------------------------------------------------------------
// V transpose + convert + per-64-group column interleave:
//   position c in a 64-key group holds source key  t*16 + j  where
//   c < 32 : t = c&1,      j = c>>1
//   c >= 32: t = 2+(c&1),  j = (c-32)>>1
// This matches the packed-pair LDS layout the attention kernel writes for P,
// so O += P@V stays correct with contiguous B-operand loads from Vt.
// ---------------------------------------------------------------------------
__global__ __launch_bounds__(256) void vtrans_kernel(
    const float* __restrict__ V, _Float16* __restrict__ Vt)
{
    int idx = blockIdx.x * 256 + threadIdx.x;   // b*D*G + d*G + kpos (kpos fastest)
    int kpos = idx & (G_ - 1);
    int bd   = idx >> 11;
    int d    = bd & (D_ - 1);
    int b    = bd >> 6;
    int g    = kpos >> 6;          // 64-key group
    int c    = kpos & 63;          // column within group
    int t    = ((c >> 4) & 2) | (c & 1);      // c<32 -> 0/1, c>=32 -> 2/3
    int j    = (c & 31) >> 1;
    int key  = (g << 6) + t * 16 + j;
    Vt[idx] = (_Float16)V[((size_t)b * G_ + key) * D_ + d];
}

// ---------------------------------------------------------------------------
// Flash-attention with post-softmax multiplicative gate M, base-2 softmax.
// Block = 2 waves, split-K: wave w handles keys [w*1024, w*1024+1024) for one
// 16-query tile, 64 keys per step; partials merged through LDS at the end.
// ---------------------------------------------------------------------------
__global__ __launch_bounds__(64) void attn_kernel(
    const _Float16* __restrict__ Qf,   // [B, G, D] f16, pre-scaled by log2(e)/sqrt(D)
    const _Float16* __restrict__ Kf,   // [B, G, D] f16
    const _Float16* __restrict__ Vt,   // [B, D, G] f16, column-interleaved per 64
    const float*    __restrict__ Msk,  // [B, G, G] f32 (streamed once)
    float*          __restrict__ Out)  // [B, G, D] f32
{
    __shared__ __align__(16) _Float16 pbuf[2][16][72];  // per-wave 16q x 64k (+pad)
    __shared__ __align__(16) float accbuf[16][68];      // wave1 O partial
    __shared__ float mbuf[16], lbuf[16];                // wave1 m / l partials
    const int wave = threadIdx.x >> 5;
    const int lane = threadIdx.x & 31;
    const int hs   = lane >> 4;      // lane half (0/1)
    const int l16  = lane & 15;
    const int b    = blockIdx.x >> 7;               // G/16 = 128 q-tiles per batch
    const int qb   = (blockIdx.x & 127) << 4;

    // A-operand (Q rows) fixed for the whole key loop.
    const _Float16* Qrow = Qf + ((size_t)b * G_ + qb + l16) * D_;
    const v16h aq0 = load16(Qrow + hs * 16,      Qrow + hs * 16 + 8);   // d 0..31
    const v16h aq1 = load16(Qrow + 32 + hs * 16, Qrow + 40 + hs * 16);   // d 32..63

    v8f acc0 = {}, acc1 = {}, acc2 = {}, acc3 = {};   // O: 16q x 64d (4 C tiles)
    float mrun[8], lrun[8];
    #pragma unroll
    for (int r = 0; r < 8; ++r) { mrun[r] = -__builtin_inff(); lrun[r] = 0.f; }

    const _Float16* Kb = Kf + (size_t)b * G_ * D_;
    const _Float16* Vb = Vt + (size_t)b * D_ * G_;
    const float*    Mb = Msk + ((size_t)b * G_ + qb) * G_;

    const int k0 = wave * (G_ / 2);
    #pragma unroll 1
    for (int kb = k0; kb < k0 + G_ / 2; kb += 64) {
        // ---- S = Q K^T : four 16x16 tiles (keys kb+16t), paired by A-operand
        v8f s0 = {}, s1 = {}, s2 = {}, s3 = {};
        {
            const _Float16* kp = Kb + (size_t)(kb + l16) * D_ + hs * 16;
            v16h bk;
            bk = load16(kp,          kp + 8);          s0 = WMMA_F16(aq0, bk, s0);
            bk = load16(kp + 1024,   kp + 1032);       s1 = WMMA_F16(aq0, bk, s1);
            bk = load16(kp + 2048,   kp + 2056);       s2 = WMMA_F16(aq0, bk, s2);
            bk = load16(kp + 3072,   kp + 3080);       s3 = WMMA_F16(aq0, bk, s3);
            bk = load16(kp + 32,     kp + 40);         s0 = WMMA_F16(aq1, bk, s0);
            bk = load16(kp + 1056,   kp + 1064);       s1 = WMMA_F16(aq1, bk, s1);
            bk = load16(kp + 2080,   kp + 2088);       s2 = WMMA_F16(aq1, bk, s2);
            bk = load16(kp + 3104,   kp + 3112);       s3 = WMMA_F16(aq1, bk, s3);
        }

        // ---- gate mask loads (the single HBM-streamed tensor) + next-step prefetch
        float msk0[8], msk1[8], msk2[8], msk3[8];
        #pragma unroll
        for (int r = 0; r < 8; ++r) {
            const float* mrow = Mb + (size_t)(hs * 8 + r) * G_ + kb;
            msk0[r] = mrow[l16];
            msk1[r] = mrow[16 + l16];
            msk2[r] = mrow[32 + l16];
            msk3[r] = mrow[48 + l16];
            __builtin_prefetch(mrow + 64 + l16, 0, 0);
        }

        // ---- online softmax, base 2; one max-reduce per row per 64 keys
        #pragma unroll
        for (int r = 0; r < 8; ++r) {
            float x  = red_max16(fmaxf(fmaxf(s0[r], s1[r]), fmaxf(s2[r], s3[r])));
            float mn = fmaxf(mrun[r], x);
            float sc = __builtin_amdgcn_exp2f(mrun[r] - mn);
            mrun[r] = mn;
            float p0 = __builtin_amdgcn_exp2f(s0[r] - mn);
            float p1 = __builtin_amdgcn_exp2f(s1[r] - mn);
            float p2 = __builtin_amdgcn_exp2f(s2[r] - mn);
            float p3 = __builtin_amdgcn_exp2f(s3[r] - mn);
            lrun[r] = lrun[r] * sc + ((p0 + p1) + (p2 + p3));  // lane-local partial
            acc0[r] *= sc; acc1[r] *= sc; acc2[r] *= sc; acc3[r] *= sc;
            // packed-pair store matching Vt's column interleave:
            // cols 2*l16 / 2*l16+1 <- tile0/tile1; cols 32+2*l16 / +1 <- tile2/tile3
            v2h pk01 = __builtin_amdgcn_cvt_pkrtz(p0 * msk0[r], p1 * msk1[r]);
            v2h pk23 = __builtin_amdgcn_cvt_pkrtz(p2 * msk2[r], p3 * msk3[r]);
            *(v2h*)&pbuf[wave][hs * 8 + r][2 * l16]      = pk01;
            *(v2h*)&pbuf[wave][hs * 8 + r][32 + 2 * l16] = pk23;
        }
        asm volatile("s_wait_dscnt 0x0" ::: "memory");  // LDS C-layout -> A-layout

        // P as two A operands (interleaved cols 0..31 and 32..63 of this step)
        const v16h pa0 = load16(&pbuf[wave][l16][hs * 16],      &pbuf[wave][l16][hs * 16 + 8]);
        const v16h pa1 = load16(&pbuf[wave][l16][32 + hs * 16], &pbuf[wave][l16][40 + hs * 16]);

        // ---- O += P @ V  (B from interleaved Vt: lane N=l16 -> d, K = hs*16+{0..15})
        const _Float16* vp = Vb + (size_t)l16 * G_ + kb + hs * 16;
        v16h bv;
        bv = load16(vp,                 vp + 8);                 acc0 = WMMA_F16(pa0, bv, acc0);
        bv = load16(vp + 16 * G_,       vp + 16 * G_ + 8);       acc1 = WMMA_F16(pa0, bv, acc1);
        bv = load16(vp + 32 * G_,       vp + 32 * G_ + 8);       acc2 = WMMA_F16(pa0, bv, acc2);
        bv = load16(vp + 48 * G_,       vp + 48 * G_ + 8);       acc3 = WMMA_F16(pa0, bv, acc3);
        bv = load16(vp + 32,            vp + 40);                acc0 = WMMA_F16(pa1, bv, acc0);
        bv = load16(vp + 16 * G_ + 32,  vp + 16 * G_ + 40);      acc1 = WMMA_F16(pa1, bv, acc1);
        bv = load16(vp + 32 * G_ + 32,  vp + 32 * G_ + 40);      acc2 = WMMA_F16(pa1, bv, acc2);
        bv = load16(vp + 48 * G_ + 32,  vp + 48 * G_ + 40);      acc3 = WMMA_F16(pa1, bv, acc3);
    }

    // ---- finalize lane-local denominators (single butterfly per row)
    #pragma unroll
    for (int r = 0; r < 8; ++r) lrun[r] = red_sum16(lrun[r]);

    // ---- split-K merge: wave1 publishes partials, wave0 combines + writes
    if (wave == 1) {
        #pragma unroll
        for (int r = 0; r < 8; ++r) {
            const int row = hs * 8 + r;
            accbuf[row][l16]      = acc0[r];
            accbuf[row][16 + l16] = acc1[r];
            accbuf[row][32 + l16] = acc2[r];
            accbuf[row][48 + l16] = acc3[r];
            mbuf[row] = mrun[r];   // row-uniform; redundant same-value writes OK
            lbuf[row] = lrun[r];
        }
    }
    __syncthreads();
    if (wave == 0) {
        #pragma unroll
        for (int r = 0; r < 8; ++r) {
            const int row = hs * 8 + r;
            float m1 = mbuf[row], l1 = lbuf[row];
            float m  = fmaxf(mrun[r], m1);
            float f0 = __builtin_amdgcn_exp2f(mrun[r] - m);
            float f1 = __builtin_amdgcn_exp2f(m1 - m);
            float inv = 1.0f / (lrun[r] * f0 + l1 * f1);
            float* op = Out + ((size_t)b * G_ + qb + row) * D_;
            op[l16]      = (acc0[r] * f0 + accbuf[row][l16]      * f1) * inv;
            op[16 + l16] = (acc1[r] * f0 + accbuf[row][16 + l16] * f1) * inv;
            op[32 + l16] = (acc2[r] * f0 + accbuf[row][32 + l16] * f1) * inv;
            op[48 + l16] = (acc3[r] * f0 + accbuf[row][48 + l16] * f1) * inv;
        }
    }
}

extern "C" void kernel_launch(void* const* d_in, const int* in_sizes, int n_in,
                              void* d_out, int out_size, void* d_ws, size_t ws_size,
                              hipStream_t stream) {
    (void)in_sizes; (void)n_in; (void)out_size; (void)ws_size;
    const float* Qg  = (const float*)d_in[0];
    const float* Kg  = (const float*)d_in[1];
    const float* Qe  = (const float*)d_in[2];
    const float* Ke  = (const float*)d_in[3];
    const float* Ve  = (const float*)d_in[4];
    const float* M   = (const float*)d_in[5];
    const float* WKw = (const float*)d_in[6];
    const float* WKb = (const float*)d_in[7];
    const float* WQw = (const float*)d_in[8];
    const float* WQb = (const float*)d_in[9];
    float* Out = (float*)d_out;

    const size_t N = (size_t)B_ * G_ * D_;      // 1M elements per tensor
    _Float16* Qf = (_Float16*)d_ws;             // 2 MB
    _Float16* Kf = Qf + N;                      // 2 MB
    _Float16* Vt = Kf + N;                      // 2 MB

    // Q pre-scale: (1/sqrt(D)) * log2(e) so softmax runs in base 2.
    const float qscale = 0.125f * 1.44269504088896340736f;
    proj_kernel<<<N / 256, 256, 0, stream>>>(Qg, Qe, WQw, WQb, Qf, qscale);
    proj_kernel<<<N / 256, 256, 0, stream>>>(Kg, Ke, WKw, WKb, Kf, 1.0f);
    vtrans_kernel<<<N / 256, 256, 0, stream>>>(Ve, Vt);

    // one block per 16-query tile, 2 split-K waves per block
    attn_kernel<<<B_ * G_ / 16, 64, 0, stream>>>(Qf, Kf, Vt, M, Out);
}
